// SocialVAE_5222680232339
// MI455X (gfx1250) — matrix operand order repaired and checked by reading, not compiled, not run on hardware
//
#include <hip/hip_runtime.h>

typedef _Float16 half_t;
typedef __attribute__((ext_vector_type(16))) _Float16 v16h;
typedef __attribute__((ext_vector_type(8)))  _Float16 v8h;
typedef __attribute__((ext_vector_type(8)))  float    v8f;

// ---------------------------------------------------------------------------
// CDNA5 async copy: global -> LDS, 16 bytes per lane, tracked by ASYNCcnt
// ---------------------------------------------------------------------------
__device__ __forceinline__ void async_ld16(unsigned lds_off, const half_t* g) {
    asm volatile("global_load_async_to_lds_b128 %0, %1, off"
                 :: "v"(lds_off), "v"(g) : "memory");
}
__device__ __forceinline__ void wait_async0() {
    asm volatile("s_wait_asynccnt 0x0" ::: "memory");
}
// ASYNCcnt retires in order: <=3 means the 3 older copies (current chunk) are
// done while the 3 just-issued prefetch copies remain in flight.
__device__ __forceinline__ void wait_async3() {
    asm volatile("s_wait_asynccnt 0x3" ::: "memory");
}

// ---------------------------------------------------------------------------
// Weight repack: f32 [K][N] row-major  ->  f16 [N][K]  (transposed, WMMA-B friendly)
// ---------------------------------------------------------------------------
__global__ void pack_wT(const float* __restrict__ w, half_t* __restrict__ wT, int K, int N) {
    int i = blockIdx.x * blockDim.x + threadIdx.x;
    if (i >= K * N) return;
    int k = i / N, n = i % N;
    wT[(long)n * K + k] = (half_t)w[i];
}

// ---------------------------------------------------------------------------
// Generic tiny-K linear layer (K = 2/3/4), f32 in (strided rows) -> f16 out
// ---------------------------------------------------------------------------
__global__ void lin_small(const float* __restrict__ in, int istride,
                          const float* __restrict__ w, const float* __restrict__ b,
                          half_t* __restrict__ out, int M, int K, int N, int act) {
    int i = blockIdx.x * blockDim.x + threadIdx.x;
    if (i >= M * N) return;
    int r = i / N, c = i % N;
    float acc = b[c];
    for (int k = 0; k < K; ++k) acc += in[(long)r * istride + k] * w[k * N + c];
    if (act) acc = fminf(fmaxf(acc, 0.f), 6.f);
    out[(long)r * N + c] = (half_t)acc;
}

// ---------------------------------------------------------------------------
// Shared epilogue for WMMA GEMMs
// ---------------------------------------------------------------------------
__device__ __forceinline__ void store_tile(v8f acc, long rowb, int N, int nidx,
                                           float bv, int act,
                                           float* outF, half_t* outH) {
#pragma unroll
    for (int r = 0; r < 8; ++r) {
        float val = acc[r] + bv;
        if (act == 1) val = fminf(fmaxf(val, 0.f), 6.f);
        long o = (rowb + r) * (long)N + nidx;
        if (outF) outF[o] = val;
        if (outH) outH[o] = (half_t)val;
    }
}

// ---------------------------------------------------------------------------
// Double-buffered LDS-staged WMMA GEMM:  out[M,N] = act(A[M,K] @ WT[N,K]^T + b)
// Block = 256 thr (8 waves), block tile 128x64. Per 32-wide K chunk the block
// async-copies A(128x32, 8KB) + B(64x32, 4KB) into one of two LDS buffers
// (global_load_async_to_lds_b128 / ASYNCcnt); the NEXT chunk is prefetched
// into the alternate buffer before s_wait_asynccnt 3, so global latency
// overlaps the 4 v_wmma_f32_16x16x32_f16 per wave. A shared by 4 waves, B by
// 2 -> ~43 flop/byte of global traffic.  Requires M%128==0, N%64==0, K%32==0.
// ---------------------------------------------------------------------------
__global__ void __launch_bounds__(256)
gemm_wmma_lds(const half_t* __restrict__ A, const half_t* __restrict__ WT,
              const float* __restrict__ bias,
              float* __restrict__ outF, half_t* __restrict__ outH,
              int M, int K, int N, int act) {
    __shared__ __align__(16) half_t sA[2][128 * 32];
    __shared__ __align__(16) half_t sB[2][64 * 32];
    const int t = threadIdx.x;
    const int lane = t & 31;
    const int wave = t >> 5;
    const int gn = N >> 6;
    const int bm = blockIdx.x / gn;
    const int bn = blockIdx.x % gn;
    const long row0 = (long)bm * 128;
    const int col0 = bn * 64;
    const int wm = wave >> 2;          // 0..1 : 64-row half of block tile
    const int wn = wave & 3;           // 0..3 : 16-col slice of block tile
    const int hi = lane >> 4, lo = lane & 15;

    // staging: thread t moves elements [t*8, t*8+8) and +2048 of the 128x32 A
    // chunk, and [t*8, t*8+8) of the 64x32 B chunk
    const int e0 = t * 8;
    const int e1 = e0 + 2048;
    const half_t* gA0 = A + (row0 + (e0 >> 5)) * (long)K + (e0 & 31);
    const half_t* gA1 = A + (row0 + (e1 >> 5)) * (long)K + (e1 & 31);
    const half_t* gB  = WT + (long)(col0 + (e0 >> 5)) * K + (e0 & 31);
    const unsigned lA0b0 = (unsigned)(size_t)(void*)(&sA[0][e0]);
    const unsigned lA1b0 = (unsigned)(size_t)(void*)(&sA[0][e1]);
    const unsigned lBb0  = (unsigned)(size_t)(void*)(&sB[0][e0]);
    const unsigned lA0b1 = (unsigned)(size_t)(void*)(&sA[1][e0]);
    const unsigned lA1b1 = (unsigned)(size_t)(void*)(&sA[1][e1]);
    const unsigned lBb1  = (unsigned)(size_t)(void*)(&sB[1][e0]);

    auto prefetch = [&](int chunk, int buf) {
        const int kc = chunk << 5;
        async_ld16(buf ? lA0b1 : lA0b0, gA0 + kc);
        async_ld16(buf ? lA1b1 : lA1b0, gA1 + kc);
        async_ld16(buf ? lBb1  : lBb0,  gB  + kc);
    };

    const int nch = K >> 5;
    prefetch(0, 0);

    v8f acc0 = {}, acc1 = {}, acc2 = {}, acc3 = {};
    for (int k = 0; k < nch; ++k) {
        const int cur = k & 1;
        // separates iter k-1 fragment reads of buf (k+1)&1 from the prefetch
        // writes into that same buffer below
        __syncthreads();
        const bool more = (k + 1 < nch);
        if (more) { prefetch(k + 1, cur ^ 1); wait_async3(); }
        else      { wait_async0(); }
        __syncthreads();           // chunk k's data visible to all 8 waves

        const half_t* fA = &sA[cur][(wm * 64 + lo) * 32 + hi * 8];
        const half_t* fB = &sB[cur][(wn * 16 + lo) * 32 + hi * 8];
        union { v16h v; v8h h[2]; } b, a0, a1, a2, a3;
        b.h[0] = *(const v8h*)(fB);
        b.h[1] = *(const v8h*)(fB + 16);
        const half_t* p = fA;
        a0.h[0] = *(const v8h*)(p);  a0.h[1] = *(const v8h*)(p + 16);  p += 16 * 32;
        a1.h[0] = *(const v8h*)(p);  a1.h[1] = *(const v8h*)(p + 16);  p += 16 * 32;
        a2.h[0] = *(const v8h*)(p);  a2.h[1] = *(const v8h*)(p + 16);  p += 16 * 32;
        a3.h[0] = *(const v8h*)(p);  a3.h[1] = *(const v8h*)(p + 16);
        acc0 = __builtin_amdgcn_wmma_f32_16x16x32_f16(false, a0.v, false, b.v, (short)0, acc0, false, false);
        acc1 = __builtin_amdgcn_wmma_f32_16x16x32_f16(false, a1.v, false, b.v, (short)0, acc1, false, false);
        acc2 = __builtin_amdgcn_wmma_f32_16x16x32_f16(false, a2.v, false, b.v, (short)0, acc2, false, false);
        acc3 = __builtin_amdgcn_wmma_f32_16x16x32_f16(false, a3.v, false, b.v, (short)0, acc3, false, false);
    }
    const int nidx = col0 + wn * 16 + lo;
    const float bv = bias ? bias[nidx] : 0.f;
    const long rb = row0 + wm * 64 + hi * 8;   // C: lanes 16-31 hold rows M=r+8
    store_tile(acc0, rb,      N, nidx, bv, act, outF, outH);
    store_tile(acc1, rb + 16, N, nidx, bv, act, outF, outH);
    store_tile(acc2, rb + 32, N, nidx, bv, act, outF, outH);
    store_tile(acc3, rb + 48, N, nidx, bv, act, outF, outH);
}

// ---------------------------------------------------------------------------
// Direct-load WMMA GEMM (fallback for N%64!=0: the N=32 p_z projections).
// One wave = 64x16 C tile, fragments straight from global.
// Requires M%64==0, N%16==0, K%32==0.
// ---------------------------------------------------------------------------
__global__ void __launch_bounds__(256)
gemm_wmma(const half_t* __restrict__ A, const half_t* __restrict__ WT,
          const float* __restrict__ bias,
          float* __restrict__ outF, half_t* __restrict__ outH,
          int M, int K, int N, int act) {
    const int lane = threadIdx.x & 31;
    const int wave = threadIdx.x >> 5;
    const int tiles_n = N >> 4;
    const long total = (long)(M >> 6) * tiles_n;
    long wid = (long)blockIdx.x * 8 + wave;
    if (wid >= total) return;                  // wave-uniform: EXEC stays all-ones
    const int tm = (int)(wid / tiles_n);
    const int tn = (int)(wid % tiles_n);
    const long row0 = (long)tm * 64;
    const int col0 = tn * 16;
    const int hi = lane >> 4;
    const int lo = lane & 15;

    const half_t* ap = A + (row0 + lo) * (long)K + hi * 8;
    const half_t* bp = WT + (long)(col0 + lo) * K + hi * 8;

    v8f acc0 = {}, acc1 = {}, acc2 = {}, acc3 = {};
    for (int kc = 0; kc < K; kc += 32) {
        union { v16h v; v8h h[2]; } b, a0, a1, a2, a3;
        b.h[0] = *(const v8h*)(bp + kc);
        b.h[1] = *(const v8h*)(bp + kc + 16);
        const half_t* p = ap + kc;
        a0.h[0] = *(const v8h*)(p);  a0.h[1] = *(const v8h*)(p + 16);  p += (long)16 * K;
        a1.h[0] = *(const v8h*)(p);  a1.h[1] = *(const v8h*)(p + 16);  p += (long)16 * K;
        a2.h[0] = *(const v8h*)(p);  a2.h[1] = *(const v8h*)(p + 16);  p += (long)16 * K;
        a3.h[0] = *(const v8h*)(p);  a3.h[1] = *(const v8h*)(p + 16);
        acc0 = __builtin_amdgcn_wmma_f32_16x16x32_f16(false, a0.v, false, b.v, (short)0, acc0, false, false);
        acc1 = __builtin_amdgcn_wmma_f32_16x16x32_f16(false, a1.v, false, b.v, (short)0, acc1, false, false);
        acc2 = __builtin_amdgcn_wmma_f32_16x16x32_f16(false, a2.v, false, b.v, (short)0, acc2, false, false);
        acc3 = __builtin_amdgcn_wmma_f32_16x16x32_f16(false, a3.v, false, b.v, (short)0, acc3, false, false);
    }
    const float bv = bias ? bias[col0 + lo] : 0.f;
    const long rb = row0 + hi * 8;
    const int nidx = col0 + lo;
    store_tile(acc0, rb,      N, nidx, bv, act, outF, outH);
    store_tile(acc1, rb + 16, N, nidx, bv, act, outF, outH);
    store_tile(acc2, rb + 32, N, nidx, bv, act, outF, outH);
    store_tile(acc3, rb + 48, N, nidx, bv, act, outF, outH);
}

// ---------------------------------------------------------------------------
// Neighbor geometry (dist / mpd / ttc) + first layers of embed_k (3->256) and
// embed_n (4->64), fused; one thread per (agent, neighbor) row for one timestep
// ---------------------------------------------------------------------------
__global__ void geom_embed0(const float* __restrict__ xl, const float* __restrict__ nbl,
                            const float* __restrict__ wk0, const float* __restrict__ bk0,
                            const float* __restrict__ wn0, const float* __restrict__ bn0,
                            half_t* __restrict__ k0, half_t* __restrict__ n0,
                            unsigned char* __restrict__ mask) {
    int r = blockIdx.x * blockDim.x + threadIdx.x;
    if (r >= 65536) return;
    int n = r >> 6;
    const float* xr = xl + n * 6;
    const float* nb = nbl + (long)r * 6;
    float dpx = nb[0] - xr[0], dpy = nb[1] - xr[1];
    float dvx = nb[2] - xr[2], dvy = nb[3] - xr[3];
    float dist = sqrtf(dpx * dpx + dpy * dpy);
    mask[r] = (dist <= 2.0f) ? 1 : 0;
    float dot = dpx * dvx + dpy * dvy;
    float tau = -dot / (dvx * dvx + dvy * dvy + 1e-6f);
    tau = fminf(fmaxf(tau, 0.f), 7.f);
    float mx = dpx + tau * dvx, my = dpy + tau * dvy;
    float mpd = sqrtf(mx * mx + my * my);
    for (int c = 0; c < 256; ++c) {
        float a = dist * wk0[c] + mpd * wk0[256 + c] + tau * wk0[512 + c] + bk0[c];
        a = fminf(fmaxf(a, 0.f), 6.f);
        k0[(long)r * 256 + c] = (half_t)a;
    }
    for (int c = 0; c < 64; ++c) {
        float a = dpx * wn0[c] + dpy * wn0[64 + c] + dvx * wn0[128 + c] + dvy * wn0[192 + c] + bn0[c];
        a = fminf(fmaxf(a, 0.f), 6.f);
        n0[(long)r * 64 + c] = (half_t)a;
    }
}

// ---------------------------------------------------------------------------
// GATv2-style attention: scores = leaky_relu(k.q), masked softmax over 64
// neighbors, value aggregation, and assembly of GRU input concat(s, n_agg).
// One 128-thread block per agent.
// ---------------------------------------------------------------------------
__global__ void attn_kernel(const half_t* __restrict__ kf, const half_t* __restrict__ q,
                            const half_t* __restrict__ v, const unsigned char* __restrict__ mask,
                            const half_t* __restrict__ s_l, half_t* __restrict__ xin) {
    __shared__ float e[64];
    __shared__ float att[64];
    int n = blockIdx.x, t = threadIdx.x;
    if (t < 64) {
        float acc = 0.f;
        const half_t* kr = kf + ((long)n * 64 + t) * 256;
        const half_t* qr = q + (long)n * 256;
        for (int d = 0; d < 256; ++d) acc += (float)kr[d] * (float)qr[d];
        acc = acc > 0.f ? acc : 0.2f * acc;              // leaky_relu(0.2)
        e[t] = mask[n * 64 + t] ? acc : -1e30f;
    }
    __syncthreads();
    if (t == 0) {
        int any = 0;
        float mx = -1e30f;
        for (int j = 0; j < 64; ++j) { if (mask[n * 64 + j]) any = 1; mx = fmaxf(mx, e[j]); }
        float sum = 0.f;
        for (int j = 0; j < 64; ++j) { float ex = __expf(e[j] - mx); att[j] = ex; sum += ex; }
        float inv = any ? 1.f / sum : 0.f;               // nan_to_num semantics
        for (int j = 0; j < 64; ++j) att[j] *= inv;
    }
    __syncthreads();
    {
        float acc = 0.f;
        const half_t* vb = v + (long)n * 64 * 128 + t;
        for (int j = 0; j < 64; ++j) acc += att[j] * (float)vb[j * 128];
        xin[(long)n * 256 + 128 + t] = (half_t)acc;      // n_agg
        xin[(long)n * 256 + t] = s_l[(long)n * 128 + t]; // self embed
    }
}

// ---------------------------------------------------------------------------
// GRU gate fusion (torch gate order r,z,n), biases already added by GEMMs
// ---------------------------------------------------------------------------
__global__ void gru_fuse(const float* __restrict__ gi, const float* __restrict__ gh,
                         float* __restrict__ h, half_t* __restrict__ h16) {
    int i = blockIdx.x * blockDim.x + threadIdx.x;
    if (i >= 1024 * 256) return;
    int n = i >> 8, c = i & 255;
    const float* gin = gi + (long)n * 768;
    const float* ghn = gh + (long)n * 768;
    float r = 1.f / (1.f + __expf(-(gin[c] + ghn[c])));
    float z = 1.f / (1.f + __expf(-(gin[256 + c] + ghn[256 + c])));
    float nn = tanhf(gin[512 + c] + r * ghn[512 + c]);
    float hp = h[i];
    float hn = (1.f - z) * nn + z * hp;
    h[i] = hn;
    h16[i] = (half_t)hn;
}

// ---------------------------------------------------------------------------
// z = mu + max(softplus(std_raw),1e-6)*eps  ;  zcat = concat(z, hf) (f16, K=288)
// ---------------------------------------------------------------------------
__global__ void z_concat(const float* __restrict__ mu, const float* __restrict__ sd,
                         const float* __restrict__ eps, const half_t* __restrict__ hf16,
                         float* __restrict__ z32, half_t* __restrict__ zcat) {
    int i = blockIdx.x * blockDim.x + threadIdx.x;
    if (i >= 1024 * 288) return;
    int n = i / 288, c = i % 288;
    if (c < 32) {
        float sr = sd[n * 32 + c];
        float sp = sr > 20.f ? sr : log1pf(__expf(sr));
        sp = fmaxf(sp, 1e-6f);
        float zv = mu[n * 32 + c] + sp * eps[n * 32 + c];
        z32[n * 32 + c] = zv;
        zcat[i] = (half_t)zv;
    } else {
        zcat[i] = hf16[(long)n * 256 + (c - 32)];
    }
}

// dec.mu: 256 -> 2 (tiny, VALU)
__global__ void dec_mu(const half_t* __restrict__ demb, const float* __restrict__ w,
                       const float* __restrict__ b, float* __restrict__ d) {
    int i = blockIdx.x * blockDim.x + threadIdx.x;
    if (i >= 2048) return;
    int n = i >> 1, c = i & 1;
    float acc = b[c];
    const half_t* dr = demb + (long)n * 256;
    for (int k = 0; k < 256; ++k) acc += (float)dr[k] * w[k * 2 + c];
    d[i] = acc;
}

// embed_zd: concat(z(32), d(2)) -> 32 (relu6) -> 32 ; one 32-thread block/agent
__global__ void zd_mlp(const float* __restrict__ z, const float* __restrict__ d,
                       const float* __restrict__ w0, const float* __restrict__ b0,
                       const float* __restrict__ w1, const float* __restrict__ b1,
                       half_t* __restrict__ zd) {
    __shared__ float t[32];
    int n = blockIdx.x, j = threadIdx.x;
    float acc = b0[j];
    for (int i = 0; i < 32; ++i) acc += z[n * 32 + i] * w0[i * 32 + j];
    acc += d[n * 2 + 0] * w0[32 * 32 + j] + d[n * 2 + 1] * w0[33 * 32 + j];
    t[j] = fminf(fmaxf(acc, 0.f), 6.f);
    __syncthreads();
    float a2 = b1[j];
    for (int i = 0; i < 32; ++i) a2 += t[i] * w1[i * 32 + j];
    zd[(long)n * 32 + j] = (half_t)a2;
}

// y_pred = cumsum(ds, axis=0) + x[-1,:,:2]
__global__ void cumsum_out(const float* __restrict__ ds, const float* __restrict__ x,
                           float* __restrict__ out) {
    int i = blockIdx.x * blockDim.x + threadIdx.x;
    if (i >= 2048) return;
    int n = i >> 1, c = i & 1;
    float acc = x[((long)8 * 1024 + n) * 6 + c];
    for (int h = 0; h < 12; ++h) {
        acc += ds[(long)h * 2048 + i];
        out[(long)h * 2048 + i] = acc;
    }
}

// ---------------------------------------------------------------------------
// Host orchestration
// ---------------------------------------------------------------------------
extern "C" void kernel_launch(void* const* d_in, const int* in_sizes, int n_in,
                              void* d_out, int out_size, void* d_ws, size_t ws_size,
                              hipStream_t stream) {
    (void)in_sizes; (void)n_in; (void)out_size; (void)ws_size;
    const float* x   = (const float*)d_in[0];   // (9,1024,6)
    const float* nb  = (const float*)d_in[1];   // (9,1024,64,6)
    const float* eps = (const float*)d_in[2];   // (12,1024,32)
    auto P = [&](int i) { return (const float*)d_in[i]; };
    // params pytree leaves (alphabetical dict keys):
    //  3: dec.embed[0].b  4: dec.embed[0].w(288x256)  5: dec.embed[1].b  6: dec.embed[1].w
    //  7: dec.mu.b  8: dec.mu.w(256x2)
    //  9..14: embed_k b0,w0(3x256),b1,w1,b2,w2   15..20: embed_n b0,w0(4x64),b1,w1(64x128),b2,w2(128x128)
    // 21..26: embed_q b0,w0,b1,w1,b2,w2          27..30: embed_s b0,w0(4x64),b1,w1(64x128)
    // 31..34: embed_zd b0,w0(34x32),b1,w1(32x32)
    // 35..40: p_z embed.b,embed.w, mu.b,mu.w(256x32), std.b,std.w
    // 41..44: rnn_fx bh,bi,wh(256x768),wi(256x768)
    // 45..50: rnn_fx_init b0,w0(2x256),b1,w1,b2,w2
    // 51..54: rnn_fy bh,bi,wh(256x768),wi(32x768)
    // 55..58: rnn_fy_init b0,w0,b1,w1

    char* base = (char*)d_ws;
    size_t off = 0;
    auto alloc = [&](size_t bytes) -> void* {
        void* p = base + off;
        off = (off + bytes + 255) & ~(size_t)255;
        return p;
    };
    auto aH = [&](size_t elems) { return (half_t*)alloc(elems * 2); };
    auto aF = [&](size_t elems) { return (float*)alloc(elems * 4); };

    // f16 transposed weights
    half_t* wk1T = aH(65536);  half_t* wk2T = aH(65536);
    half_t* wn1T = aH(8192);   half_t* wn2T = aH(16384);
    half_t* ws1T = aH(8192);
    half_t* wq0T = aH(65536);  half_t* wq1T = aH(65536);  half_t* wq2T = aH(65536);
    half_t* wfxi1T = aH(65536); half_t* wfxi2T = aH(65536);
    half_t* wfy0T = aH(65536); half_t* wfy1T = aH(65536);
    half_t* wfxiT = aH(196608); half_t* wfxhT = aH(196608);
    half_t* wpzT = aH(65536);  half_t* wpzmuT = aH(8192);  half_t* wpzsdT = aH(8192);
    half_t* wd0T = aH(73728);  half_t* wd1T = aH(65536);
    half_t* wfyiT = aH(24576); half_t* wfyhT = aH(196608);

    // activations
    half_t* s0tmp = aH(9216 * 64);
    half_t* s16   = aH((size_t)9216 * 128);
    half_t* k0buf = aH((size_t)65536 * 256);   // layer0 out, reused for final k
    half_t* k1buf = aH((size_t)65536 * 256);
    half_t* n0buf = aH((size_t)65536 * 64);
    half_t* n1buf = aH((size_t)65536 * 128);
    half_t* vfin  = aH((size_t)65536 * 128);
    unsigned char* maskb = (unsigned char*)alloc(65536);
    float*  h32  = aF(1024 * 256);  half_t* h16  = aH(1024 * 256);
    half_t* ta16 = aH(1024 * 256);  half_t* tb16 = aH(1024 * 256);
    half_t* q16  = aH(1024 * 256);  half_t* xin16 = aH(1024 * 256);
    float*  gi32 = aF(1024 * 768);  float*  gh32 = aF(1024 * 768);
    float*  hf32 = aF(1024 * 256);  half_t* hf16 = aH(1024 * 256);
    float*  mu32 = aF(1024 * 32);   float*  sd32 = aF(1024 * 32);
    float*  z32  = aF(1024 * 32);
    half_t* zcat16 = aH(1024 * 288);
    half_t* zd16 = aH(1024 * 32);
    float*  dsbuf = aF(12 * 1024 * 2);

    auto packW = [&](int idx, half_t* dst, int K, int N) {
        int tot = K * N;
        pack_wT<<<(tot + 255) / 256, 256, 0, stream>>>(P(idx), dst, K, N);
    };
    auto gemm = [&](const half_t* A, const half_t* WT, const float* bias,
                    float* outF, half_t* outH, int M, int K, int N, int act) {
        if ((M % 128 == 0) && (N % 64 == 0)) {
            int blocks = (M / 128) * (N / 64);
            gemm_wmma_lds<<<blocks, 256, 0, stream>>>(A, WT, bias, outF, outH, M, K, N, act);
        } else {
            long waves = (long)(M / 64) * (N / 16);
            int blocks = (int)((waves + 7) / 8);
            gemm_wmma<<<blocks, 256, 0, stream>>>(A, WT, bias, outF, outH, M, K, N, act);
        }
    };

    // --- pack weights ---
    packW(12, wk1T, 256, 256);  packW(14, wk2T, 256, 256);
    packW(18, wn1T, 64, 128);   packW(20, wn2T, 128, 128);
    packW(30, ws1T, 64, 128);
    packW(22, wq0T, 256, 256);  packW(24, wq1T, 256, 256);  packW(26, wq2T, 256, 256);
    packW(48, wfxi1T, 256, 256); packW(50, wfxi2T, 256, 256);
    packW(56, wfy0T, 256, 256); packW(58, wfy1T, 256, 256);
    packW(44, wfxiT, 256, 768); packW(43, wfxhT, 256, 768);
    packW(36, wpzT, 256, 256);  packW(38, wpzmuT, 256, 32); packW(40, wpzsdT, 256, 32);
    packW(4,  wd0T, 288, 256);  packW(6,  wd1T, 256, 256);
    packW(54, wfyiT, 32, 768);  packW(53, wfyhT, 256, 768);

    // --- self embed s = embed_s(x[...,2:]) for all L*N rows ---
    lin_small<<<(9216 * 64 + 255) / 256, 256, 0, stream>>>(x + 2, 6, P(28), P(27),
                                                           s0tmp, 9216, 4, 64, 1);
    gemm(s0tmp, ws1T, P(29), nullptr, s16, 9216, 64, 128, 0);

    // --- h0 = rnn_fx_init(x[0,:,2:4]) ---
    lin_small<<<(1024 * 256 + 255) / 256, 256, 0, stream>>>(x + 2, 6, P(46), P(45),
                                                            ta16, 1024, 2, 256, 1);
    gemm(ta16, wfxi1T, P(47), nullptr, tb16, 1024, 256, 256, 1);
    gemm(tb16, wfxi2T, P(49), h32, h16, 1024, 256, 256, 0);

    // --- encoder scan over L=9 timesteps ---
    for (int l = 0; l < 9; ++l) {
        geom_embed0<<<256, 256, 0, stream>>>(x + (long)l * 1024 * 6,
                                             nb + (long)l * 1024 * 64 * 6,
                                             P(10), P(9), P(16), P(15),
                                             k0buf, n0buf, maskb);
        gemm(k0buf, wk1T, P(11), nullptr, k1buf, 65536, 256, 256, 1);
        gemm(k1buf, wk2T, P(13), nullptr, k0buf, 65536, 256, 256, 0);   // final k
        gemm(n0buf, wn1T, P(17), nullptr, n1buf, 65536, 64, 128, 1);
        gemm(n1buf, wn2T, P(19), nullptr, vfin, 65536, 128, 128, 0);    // final v
        gemm(h16,  wq0T, P(21), nullptr, ta16, 1024, 256, 256, 1);
        gemm(ta16, wq1T, P(23), nullptr, tb16, 1024, 256, 256, 1);
        gemm(tb16, wq2T, P(25), nullptr, q16, 1024, 256, 256, 0);
        attn_kernel<<<1024, 128, 0, stream>>>(k0buf, q16, vfin, maskb,
                                              s16 + (long)l * 1024 * 128, xin16);
        gemm(xin16, wfxiT, P(42), gi32, nullptr, 1024, 256, 768, 0);
        gemm(h16,   wfxhT, P(41), gh32, nullptr, 1024, 256, 768, 0);
        gru_fuse<<<1024, 256, 0, stream>>>(gi32, gh32, h32, h16);
    }

    // --- hf = rnn_fy_init(h) ---
    gemm(h16,  wfy0T, P(55), nullptr, ta16, 1024, 256, 256, 1);
    gemm(ta16, wfy1T, P(57), hf32, hf16, 1024, 256, 256, 0);

    // --- decoder scan over H=12 horizon steps ---
    for (int hs = 0; hs < 12; ++hs) {
        gemm(hf16, wpzT, P(35), nullptr, ta16, 1024, 256, 256, 1);      // p_z embed (relu6)
        gemm(ta16, wpzmuT, P(37), mu32, nullptr, 1024, 256, 32, 0);
        gemm(ta16, wpzsdT, P(39), sd32, nullptr, 1024, 256, 32, 0);
        z_concat<<<(1024 * 288 + 255) / 256, 256, 0, stream>>>(mu32, sd32,
                eps + (long)hs * 1024 * 32, hf16, z32, zcat16);
        gemm(zcat16, wd0T, P(3), nullptr, ta16, 1024, 288, 256, 1);
        gemm(ta16,   wd1T, P(5), nullptr, tb16, 1024, 256, 256, 1);
        dec_mu<<<8, 256, 0, stream>>>(tb16, P(8), P(7), dsbuf + (long)hs * 2048);
        zd_mlp<<<1024, 32, 0, stream>>>(z32, dsbuf + (long)hs * 2048,
                                        P(32), P(31), P(34), P(33), zd16);
        gemm(zd16, wfyiT, P(52), gi32, nullptr, 1024, 32, 768, 0);
        gemm(hf16, wfyhT, P(51), gh32, nullptr, 1024, 256, 768, 0);
        gru_fuse<<<1024, 256, 0, stream>>>(gi32, gh32, hf32, hf16);
    }

    cumsum_out<<<8, 256, 0, stream>>>(dsbuf, x, (float*)d_out);
}
